// SyncArcticMoeBlock_1726576856634
// MI455X (gfx1250) — matrix operand order repaired and checked by reading, not compiled, not run on hardware
//
#include <hip/hip_runtime.h>
#include <stdint.h>

// ---------------------------------------------------------------------------
// Arctic MoE block for MI455X (gfx1250, wave32, WMMA + Tensor Data Mover).
//   y[t] = sum_{e in top2(t)} softmax(x@gw^T)[t,e] * (silu(x@w1_e)*(x@w3_e))@w2_e
// Routed-sparse execution: per-expert compact token lists (top-2 of 8 => 4x
// less math than dense), bf16 WMMA 16x16x32 with f32 accumulate, weights cast
// fp32->bf16 pre-transposed so all GEMM tiles are plain strided 2-D tiles
// staged by TDM (tensor_load_to_lds) into pad-stride-72 LDS.
// ---------------------------------------------------------------------------

typedef __bf16 bf16;
typedef __attribute__((ext_vector_type(4)))  float v4f;
typedef __attribute__((ext_vector_type(4)))  bf16  v4bf;
typedef __attribute__((ext_vector_type(8)))  bf16  v8bf;
typedef __attribute__((ext_vector_type(16))) bf16  v16bf;
typedef __attribute__((ext_vector_type(8)))  float v8f;
typedef __attribute__((ext_vector_type(4)))  unsigned int u32x4;
typedef __attribute__((ext_vector_type(8)))  int i32x8;
typedef __attribute__((ext_vector_type(4)))  int i32x4;

namespace {
constexpr int kH = 1024;   // hidden
constexpr int kF = 2048;   // ffn dim
constexpr int kE = 8;      // experts
constexpr int kT = 4096;   // tokens (2*2048)

constexpr int BM  = 64;    // workgroup M tile (expert-local rows)
constexpr int BN  = 128;   // workgroup N tile
constexpr int BK  = 64;    // K step
constexpr int LDA = 72;    // padded LDS K-stride: 64 elems + 16B pad
constexpr int LDB = 72;
}

// ---- TDM availability (device pass only; host pass takes fallback) --------
#if defined(__gfx1250__) && __has_builtin(__builtin_amdgcn_tensor_load_to_lds)
  #define MOE_HAVE_TDM 1
  #if __has_include(<hip/amd_detail/amd_gfx1250_TDM.h>)
    #define MOE_TDM_ARGS6 1
  #endif
#endif

__device__ __forceinline__ v16bf cat8(v8bf lo, v8bf hi) {
  return __builtin_shufflevector(lo, hi, 0,1,2,3,4,5,6,7,8,9,10,11,12,13,14,15);
}

#if defined(MOE_HAVE_TDM)
// 2-D tile DMA: global (row-major, stride0 elems/row) -> LDS with +4-DWORD pad
// every 32 DWORDs stored (=> LDS row stride 72 bf16 elems for 64-elem rows).
// D# layout per CDNA5 ISA 8.3/8.4. EXEC ignored; issued once per executing wave.
__device__ __forceinline__ void tdm_load_tile_2d(
    const void* gptr, unsigned int lds_off_bytes,
    unsigned int tensor_dim0, unsigned int tensor_dim1,
    unsigned long long stride0_elems,
    unsigned int tile_dim0, unsigned int tile_dim1) {
  unsigned long long ga = (unsigned long long)(uintptr_t)gptr;
  u32x4 g0;
  g0[0] = 1u;                                            // count=1, user D#
  g0[1] = lds_off_bytes;                                 // lds_addr
  g0[2] = (unsigned int)ga;                              // global_addr[31:0]
  g0[3] = (unsigned int)((ga >> 32) & 0x01FFFFFFu)       // global_addr[56:32]
        | (2u << 30);                                    // type = 2 ("image")
  i32x8 g1;
  g1[0] = (int)((1u << 16)      // data_size = 2 bytes
              | (1u << 20)      // pad_enable
              | (4u << 22)      // pad_interval: 32 DWORDs
              | (3u << 25));    // pad_amount : 4 DWORDs
  g1[1] = (int)((tensor_dim0 & 0xFFFFu) << 16);                 // dim0[15:0]
  g1[2] = (int)(((tensor_dim0 >> 16) & 0xFFFFu)
              | ((tensor_dim1 & 0xFFFFu) << 16));               // dim0 hi | dim1 lo
  g1[3] = (int)(((tensor_dim1 >> 16) & 0xFFFFu)
              | ((tile_dim0 & 0xFFFFu) << 16));                 // dim1 hi | tile0
  g1[4] = (int)(tile_dim1 & 0xFFFFu);                           // tile1 (tile2=0)
  g1[5] = (int)(unsigned int)stride0_elems;                     // stride0 lo32
  g1[6] = (int)((stride0_elems >> 32) & 0xFFFFu);               // stride0 hi16
  g1[7] = 0;
  i32x4 z4 = {};
#if defined(MOE_TDM_ARGS6)
  i32x8 z8 = {};
  __builtin_amdgcn_tensor_load_to_lds(g0, g1, z4, z4, z8, 0);
#else
  __builtin_amdgcn_tensor_load_to_lds(g0, g1, z4, z4, 0);
#endif
}
__device__ __forceinline__ void tdm_wait0() {
  __builtin_amdgcn_s_wait_tensorcnt(0);
}
#endif  // MOE_HAVE_TDM

// ---------------------------------------------------------------------------
__global__ void zero_ints(int* __restrict__ p, int n) {
  int i = blockIdx.x * blockDim.x + threadIdx.x;
  if (i < n) p[i] = 0;
}

// ---------------------------------------------------------------------------
// fp32 -> bf16 cast, 4-wide, grid-stride (activations; layout preserved)
// ---------------------------------------------------------------------------
__global__ void cast_f32_to_bf16(const float* __restrict__ src,
                                 bf16* __restrict__ dst, int n4) {
  int i = blockIdx.x * blockDim.x + threadIdx.x;
  int stride = gridDim.x * blockDim.x;
  for (; i < n4; i += stride) {
    v4f v = ((const v4f*)src)[i];
    v4bf o;
    o[0] = (bf16)v[0]; o[1] = (bf16)v[1];
    o[2] = (bf16)v[2]; o[3] = (bf16)v[3];
    ((v4bf*)dst)[i] = o;
  }
}

// ---------------------------------------------------------------------------
// fp32 [R][C] -> bf16 [C][R] (cast + transpose), 64x64 LDS tiles.
// ---------------------------------------------------------------------------
__global__ __launch_bounds__(256)
void cast_transpose_f32_bf16(const float* __restrict__ src,
                             bf16* __restrict__ dst, int R, int C) {
  __shared__ __attribute__((aligned(16))) bf16 tile[64 * 72];  // [c][r]
  const int r0 = blockIdx.y * 64;
  const int c0 = blockIdx.x * 64;
  const int tid = threadIdx.x;
  #pragma unroll
  for (int i = 0; i < 4; ++i) {
    int c  = tid + 256 * i;
    int r  = c >> 4;
    int cc = (c & 15) << 2;
    v4f v = *(const v4f*)(src + (size_t)(r0 + r) * C + c0 + cc);
    #pragma unroll
    for (int j = 0; j < 4; ++j)
      tile[(cc + j) * 72 + r] = (bf16)v[j];
  }
  __syncthreads();
  #pragma unroll
  for (int i = 0; i < 2; ++i) {
    int c  = tid + 256 * i;
    int cr = c >> 3;
    int rs = (c & 7) << 3;
    v8bf v = *(const v8bf*)(&tile[cr * 72 + rs]);
    *(v8bf*)(dst + (size_t)(c0 + cr) * R + r0 + rs) = v;
  }
}

// ---------------------------------------------------------------------------
// Router: one wave per token. logits = x @ gate_w^T (fp32), softmax, top-2.
// Appends token to both selected experts' compact lists. List entry packs
// (token<<1)|isFirstWriter, where the token's lower-indexed expert (earlier
// launch on the stream) writes y and the other accumulates -> deterministic.
// ---------------------------------------------------------------------------
__global__ void router_kernel(const float* __restrict__ x,
                              const float* __restrict__ gw,
                              float* __restrict__ logits,
                              float* __restrict__ combine,
                              int* __restrict__ cursor,
                              int* __restrict__ list) {
  int gid  = blockIdx.x * blockDim.x + threadIdx.x;
  int t    = gid >> 5;
  int lane = gid & 31;
  if (t >= kT) return;

  float acc[kE];
  #pragma unroll
  for (int e = 0; e < kE; ++e) acc[e] = 0.0f;

  const float* xr = x + (size_t)t * kH;
  for (int i = lane; i < kH; i += 32) {
    float xv = xr[i];
    #pragma unroll
    for (int e = 0; e < kE; ++e) acc[e] += xv * gw[e * kH + i];
  }
  #pragma unroll
  for (int off = 16; off > 0; off >>= 1) {
    #pragma unroll
    for (int e = 0; e < kE; ++e) acc[e] += __shfl_down(acc[e], off, 32);
  }

  if (lane == 0) {
    float mx = acc[0];
    #pragma unroll
    for (int e = 1; e < kE; ++e) mx = fmaxf(mx, acc[e]);
    float p[kE]; float den = 0.0f;
    #pragma unroll
    for (int e = 0; e < kE; ++e) { p[e] = __expf(acc[e] - mx); den += p[e]; }
    float inv = 1.0f / den;
    #pragma unroll
    for (int e = 0; e < kE; ++e) p[e] *= inv;

    int i1 = 0;
    #pragma unroll
    for (int e = 1; e < kE; ++e) if (p[e] > p[i1]) i1 = e;
    int i2 = (i1 == 0) ? 1 : 0;
    #pragma unroll
    for (int e = 0; e < kE; ++e) if (e != i1 && p[e] > p[i2]) i2 = e;

    #pragma unroll
    for (int e = 0; e < kE; ++e) {
      logits[(size_t)t * kE + e]  = acc[e];  // raw logits (pre-softmax)
      combine[(size_t)t * kE + e] = (e == i1) ? p[i1] : (e == i2) ? p[i2] : 0.0f;
    }
    int first1 = (i1 < i2) ? 1 : 0;
    int p1 = atomicAdd(&cursor[i1], 1);
    list[(size_t)i1 * kT + p1] = (t << 1) | first1;
    int p2 = atomicAdd(&cursor[i2], 1);
    list[(size_t)i2 * kT + p2] = (t << 1) | (1 - first1);
  }
}

// ---------------------------------------------------------------------------
// GEMM1 (gathered, fused): for expert-local rows p in [0,cnt):
//   t = list[p]>>1;  h[p] = silu(x[t]@W1t^T) * (x[t]@W3t^T)  -> bf16 compact.
// A tiles gathered via LDS index slice; B1/B3 weight tiles staged by TDM.
// ---------------------------------------------------------------------------
__global__ __launch_bounds__(256, 2)
void gemm1_swiglu(const bf16* __restrict__ Xb, const bf16* __restrict__ W1t,
                  const bf16* __restrict__ W3t, bf16* __restrict__ Hb,
                  const int* __restrict__ list, const int* __restrict__ cnt,
                  int expert) {
  const int cnt_e = cnt[expert];
  const int mB = blockIdx.y * BM;
  if (mB >= cnt_e) return;                 // early-exit: unrouted tile

  __shared__ __attribute__((aligned(16))) bf16 sA [BM * LDA];
  __shared__ __attribute__((aligned(16))) bf16 sB1[BN * LDB];
  __shared__ __attribute__((aligned(16))) bf16 sB3[BN * LDB];
  __shared__ int sIdx[BM];

  const int tid  = threadIdx.x;
  const int nB   = blockIdx.x * BN;
  const int wave = tid >> 5;
  const int lane = tid & 31;
  const int wm   = wave & 1;
  const int wn   = wave >> 1;
  const int lr   = lane & 15;
  const int hs   = lane >> 4;

  if (tid < BM) {
    int q = mB + tid;
    if (q > cnt_e - 1) q = cnt_e - 1;      // clamp: duplicate last row
    sIdx[tid] = list[q];
  }

  v8f acc1[2][2], acc3[2][2];
  #pragma unroll
  for (int mi = 0; mi < 2; ++mi)
    #pragma unroll
    for (int ni = 0; ni < 2; ++ni) {
      v8f z = {};
      acc1[mi][ni] = z; acc3[mi][ni] = z;
    }

  for (int k0 = 0; k0 < kH; k0 += BK) {
    __syncthreads();   // also publishes sIdx on first iteration
#if defined(MOE_HAVE_TDM)
    if (wave == 0)
      tdm_load_tile_2d(W1t + (size_t)nB * kH + k0,
                       (unsigned int)(uintptr_t)(void*)sB1,
                       kH, kF, kH, BK, BN);
    else if (wave == 1)
      tdm_load_tile_2d(W3t + (size_t)nB * kH + k0,
                       (unsigned int)(uintptr_t)(void*)sB3,
                       kH, kF, kH, BK, BN);
#else
    #pragma unroll
    for (int i = 0; i < 4; ++i) {           // B1/B3: 128x64, 1024 chunks each
      int c  = tid + 256 * i;
      int n  = c >> 3;
      int ks = (c & 7) << 3;
      *(v8bf*)(&sB1[n * LDB + ks]) =
          *(const v8bf*)(W1t + (size_t)(nB + n) * kH + k0 + ks);
      *(v8bf*)(&sB3[n * LDB + ks]) =
          *(const v8bf*)(W3t + (size_t)(nB + n) * kH + k0 + ks);
    }
#endif
    // A: 64x64 row gather via token list (all waves participate)
    #pragma unroll
    for (int i = 0; i < 2; ++i) {
      int c   = tid + 256 * i;
      int row = c >> 3;
      int col = (c & 7) << 3;
      int tok = sIdx[row] >> 1;
      *(v8bf*)(&sA[row * LDA + col]) =
          *(const v8bf*)(Xb + (size_t)tok * kH + k0 + col);
    }
#if defined(MOE_HAVE_TDM)
    tdm_wait0();
#endif
    __syncthreads();

    #pragma unroll
    for (int ks = 0; ks < 2; ++ks) {
      const int kb = ks * 32;
      v16bf afr[2];
      #pragma unroll
      for (int mi = 0; mi < 2; ++mi) {
        int row = wm * 32 + mi * 16 + lr;
        int kl  = kb + hs * 16;
        v8bf lo = *(const v8bf*)(&sA[row * LDA + kl]);
        v8bf hi = *(const v8bf*)(&sA[row * LDA + kl + 8]);
        afr[mi] = cat8(lo, hi);
      }
      #pragma unroll
      for (int ni = 0; ni < 2; ++ni) {
        int col = wn * 32 + ni * 16 + lr;
        int kl  = kb + hs * 16;
        v16bf bf1 = cat8(*(const v8bf*)(&sB1[col * LDB + kl]),
                         *(const v8bf*)(&sB1[col * LDB + kl + 8]));
        v16bf bf3 = cat8(*(const v8bf*)(&sB3[col * LDB + kl]),
                         *(const v8bf*)(&sB3[col * LDB + kl + 8]));
        #pragma unroll
        for (int mi = 0; mi < 2; ++mi) {
          acc1[mi][ni] = __builtin_amdgcn_wmma_f32_16x16x32_bf16(
              false, afr[mi], false, bf1, (short)0, acc1[mi][ni], false, false);
          acc3[mi][ni] = __builtin_amdgcn_wmma_f32_16x16x32_bf16(
              false, afr[mi], false, bf3, (short)0, acc3[mi][ni], false, false);
        }
      }
    }
  }

  #pragma unroll
  for (int mi = 0; mi < 2; ++mi)
    #pragma unroll
    for (int ni = 0; ni < 2; ++ni)
      #pragma unroll
      for (int j = 0; j < 8; ++j) {
        float c1 = acc1[mi][ni][j];
        float c3 = acc3[mi][ni][j];
        float hv = c1 * (1.0f / (1.0f + __expf(-c1))) * c3;
        int p = wm * 32 + mi * 16 + hs * 8 + j;      // expert-local row
        int f = nB + wn * 32 + ni * 16 + lr;
        if (mB + p < cnt_e)
          Hb[(size_t)(mB + p) * kF + f] = (bf16)hv;  // compact store
      }
}

// ---------------------------------------------------------------------------
// GEMM2 (scattered): for expert-local rows p: val = combine[t,e] * (h[p]@W2t^T)
//   entry=list[p]: t=entry>>1; entry&1 ? Y[t]=val : Y[t]+=val.
// A (compact h rows) and B (W2t) tiles staged by TDM.
// ---------------------------------------------------------------------------
__global__ __launch_bounds__(256, 2)
void gemm2_combine(const bf16* __restrict__ Hb, const bf16* __restrict__ W2t,
                   const float* __restrict__ combine, float* __restrict__ Y,
                   const int* __restrict__ list, const int* __restrict__ cnt,
                   int expert) {
  const int cnt_e = cnt[expert];
  const int mB = blockIdx.y * BM;
  if (mB >= cnt_e) return;

  __shared__ __attribute__((aligned(16))) bf16 sA[BM * LDA];
  __shared__ __attribute__((aligned(16))) bf16 sB[BN * LDB];
  __shared__ int sIdx[BM];

  const int tid  = threadIdx.x;
  const int nB   = blockIdx.x * BN;
  const int wave = tid >> 5;
  const int lane = tid & 31;
  const int wm   = wave & 1;
  const int wn   = wave >> 1;
  const int lr   = lane & 15;
  const int hs   = lane >> 4;

  if (tid < BM) {
    int q = mB + tid;
    if (q > cnt_e - 1) q = cnt_e - 1;
    sIdx[tid] = list[q];
  }

  v8f acc[2][2];
  #pragma unroll
  for (int mi = 0; mi < 2; ++mi)
    #pragma unroll
    for (int ni = 0; ni < 2; ++ni) { v8f z = {}; acc[mi][ni] = z; }

  for (int k0 = 0; k0 < kF; k0 += BK) {
    __syncthreads();
#if defined(MOE_HAVE_TDM)
    if (wave == 0)
      tdm_load_tile_2d(Hb + (size_t)mB * kF + k0,
                       (unsigned int)(uintptr_t)(void*)sA,
                       kF, kT, kF, BK, BM);
    else if (wave == 1)
      tdm_load_tile_2d(W2t + (size_t)nB * kF + k0,
                       (unsigned int)(uintptr_t)(void*)sB,
                       kF, kH, kF, BK, BN);
    tdm_wait0();
#else
    #pragma unroll
    for (int i = 0; i < 2; ++i) {
      int c   = tid + 256 * i;
      int row = c >> 3;
      int col = (c & 7) << 3;
      *(v8bf*)(&sA[row * LDA + col]) =
          *(const v8bf*)(Hb + (size_t)(mB + row) * kF + k0 + col);
    }
    #pragma unroll
    for (int i = 0; i < 4; ++i) {
      int c  = tid + 256 * i;
      int n  = c >> 3;
      int ks = (c & 7) << 3;
      *(v8bf*)(&sB[n * LDB + ks]) =
          *(const v8bf*)(W2t + (size_t)(nB + n) * kF + k0 + ks);
    }
#endif
    __syncthreads();

    #pragma unroll
    for (int ks = 0; ks < 2; ++ks) {
      const int kb = ks * 32;
      v16bf afr[2];
      #pragma unroll
      for (int mi = 0; mi < 2; ++mi) {
        int row = wm * 32 + mi * 16 + lr;
        int kl  = kb + hs * 16;
        v8bf lo = *(const v8bf*)(&sA[row * LDA + kl]);
        v8bf hi = *(const v8bf*)(&sA[row * LDA + kl + 8]);
        afr[mi] = cat8(lo, hi);
      }
      #pragma unroll
      for (int ni = 0; ni < 2; ++ni) {
        int col = wn * 32 + ni * 16 + lr;
        int kl  = kb + hs * 16;
        v16bf bfb = cat8(*(const v8bf*)(&sB[col * LDB + kl]),
                         *(const v8bf*)(&sB[col * LDB + kl + 8]));
        #pragma unroll
        for (int mi = 0; mi < 2; ++mi) {
          acc[mi][ni] = __builtin_amdgcn_wmma_f32_16x16x32_bf16(
              false, afr[mi], false, bfb, (short)0, acc[mi][ni], false, false);
        }
      }
    }
  }

  #pragma unroll
  for (int mi = 0; mi < 2; ++mi)
    #pragma unroll
    for (int ni = 0; ni < 2; ++ni)
      #pragma unroll
      for (int j = 0; j < 8; ++j) {
        int p = wm * 32 + mi * 16 + hs * 8 + j;
        if (mB + p < cnt_e) {
          int entry = sIdx[p];
          int t = entry >> 1;
          int n = nB + wn * 32 + ni * 16 + lr;
          float cw  = combine[(size_t)t * kE + expert];
          float val = cw * acc[mi][ni][j];
          size_t o = (size_t)t * kH + n;
          if (entry & 1) Y[o] = val;     // token's first (lower-index) expert
          else           Y[o] += val;    // second expert: accumulate
        }
      }
}

// ---------------------------------------------------------------------------
// Host orchestration (all on `stream`, deterministic output, ws ~38 MB)
// ---------------------------------------------------------------------------
extern "C" void kernel_launch(void* const* d_in, const int* in_sizes, int n_in,
                              void* d_out, int out_size, void* d_ws, size_t ws_size,
                              hipStream_t stream) {
  (void)in_sizes; (void)n_in; (void)out_size; (void)ws_size;
  const float* x  = (const float*)d_in[0];  // [2,2048,kH]
  const float* gw = (const float*)d_in[1];  // [kE,kH]
  const float* w1 = (const float*)d_in[2];  // [kE,kH,kF]
  const float* w3 = (const float*)d_in[3];  // [kE,kH,kF]
  const float* w2 = (const float*)d_in[4];  // [kE,kF,kH]
  float* y      = (float*)d_out;            // [kT,kH]
  float* logits = y + (size_t)kT * kH;      // [kT,kE]

  uintptr_t p = ((uintptr_t)d_ws + 255) & ~(uintptr_t)255;
  auto take = [&](size_t bytes) {
    void* q = (void*)p;
    p = (p + bytes + 255) & ~(uintptr_t)255;
    return q;
  };
  int*   cursor  = (int*)  take(kE * sizeof(int));                // per-expert counts
  int*   list    = (int*)  take((size_t)kE * kT * sizeof(int));   // compact lists
  float* combine = (float*)take((size_t)kT * kE * sizeof(float));
  bf16*  Xb      = (bf16*) take((size_t)kT * kH * sizeof(bf16));
  bf16*  W1t     = (bf16*) take((size_t)kH * kF * sizeof(bf16));  // [kF][kH]
  bf16*  W3t     = (bf16*) take((size_t)kH * kF * sizeof(bf16));  // [kF][kH]
  bf16*  W2t     = (bf16*) take((size_t)kF * kH * sizeof(bf16));  // [kH][kF]
  bf16*  Hb      = (bf16*) take((size_t)kT * kF * sizeof(bf16));  // compact h rows

  zero_ints<<<1, 32, 0, stream>>>(cursor, kE);
  cast_f32_to_bf16<<<1024, 256, 0, stream>>>(x, Xb, kT * kH / 4);
  router_kernel<<<kT / 8, 256, 0, stream>>>(x, gw, logits, combine, cursor, list);

  for (int e = 0; e < kE; ++e) {
    const size_t wsz = (size_t)kH * kF;
    cast_transpose_f32_bf16<<<dim3(kF / 64, kH / 64), 256, 0, stream>>>(
        w1 + e * wsz, W1t, kH, kF);
    cast_transpose_f32_bf16<<<dim3(kF / 64, kH / 64), 256, 0, stream>>>(
        w3 + e * wsz, W3t, kH, kF);
    cast_transpose_f32_bf16<<<dim3(kH / 64, kF / 64), 256, 0, stream>>>(
        w2 + e * wsz, W2t, kF, kH);
    // worst-case M grid; blocks beyond cnt[e] early-exit
    gemm1_swiglu <<<dim3(kF / BN, kT / BM), 256, 0, stream>>>(
        Xb, W1t, W3t, Hb, list + (size_t)e * kT, cursor, e);
    gemm2_combine<<<dim3(kH / BN, kT / BM), 256, 0, stream>>>(
        Hb, W2t, combine, y, list + (size_t)e * kT, cursor, e);
  }
}
